// _BaseGAT_94489281063
// MI455X (gfx1250) — compile-verified, matrix operand order checked by reading, not run on hardware
//
#include <hip/hip_runtime.h>

// ---------------- problem constants (match reference) ----------------
#define NN      50000
#define EE      400000
#define INDIM   128
#define HIDD    64
#define NHEADS  4
#define NCLS    16
#define NEG_SLOPE 0.2f

// ---------------- WMMA types ----------------
typedef __bf16 bf16_t;
typedef bf16_t v16bf __attribute__((ext_vector_type(16)));
typedef float  v8f   __attribute__((ext_vector_type(8)));

// ---------------- helpers ----------------
__device__ __forceinline__ unsigned short f32_to_bf16_rne(float f) {
    unsigned int u = __float_as_uint(f);
    unsigned int r = u + 0x7FFFu + ((u >> 16) & 1u);
    return (unsigned short)(r >> 16);
}

// monotone float<->uint encoding for atomicMax-based segment max
__device__ __forceinline__ unsigned int enc_f(float f) {
    unsigned int u = __float_as_uint(f);
    unsigned int m = ((unsigned int)((int)u >> 31)) | 0x80000000u;
    return u ^ m;
}
__device__ __forceinline__ float dec_f(unsigned int u) {
    unsigned int m = (u & 0x80000000u) ? 0x80000000u : 0xFFFFFFFFu;
    return __uint_as_float(u ^ m);
}
#define ENC_NEG_INF 0x007FFFFFu   // enc_f(-inf)

__device__ __forceinline__ void edge_sd(const int* __restrict__ ei, int e, int& s, int& d) {
    if (e < EE) { s = ei[e]; d = ei[EE + e]; }
    else        { s = d = e - EE; }          // appended self-loops
}

__device__ __forceinline__ float lrelu(float v) { return v > 0.0f ? v : NEG_SLOPE * v; }

// ---------------- kernels ----------------

// fp32 -> bf16 (row-major, GEMM A operand)
__global__ void k_f32_to_bf16(const float* __restrict__ src, unsigned short* __restrict__ dst, int n) {
    int t = blockIdx.x * blockDim.x + threadIdx.x;
    if (t < n) dst[t] = f32_to_bf16_rne(src[t]);
}

// pack W [K x Ncols] row-major fp32 -> Bpack[kb][Ncols][16] bf16 with Bpack[kb][n][j] = W[kb*16+j][n]
__global__ void k_pack_B(const float* __restrict__ W, unsigned short* __restrict__ Bp, int K, int Ncols) {
    int t = blockIdx.x * blockDim.x + threadIdx.x;
    int tot = (K / 16) * Ncols;
    if (t >= tot) return;
    int kb = t / Ncols, n = t % Ncols;
    unsigned short* o = Bp + ((size_t)kb * Ncols + n) * 16;
#pragma unroll
    for (int j = 0; j < 16; ++j)
        o[j] = f32_to_bf16_rne(W[(size_t)(kb * 16 + j) * Ncols + n]);
}

// one wave computes a 16 x (NT*16) strip of C = A(bf16,[M x K]) * Bpack(bf16); f32 accumulate.
// A fragment is loaded once per K-chunk and reused across the NT back-to-back WMMAs.
template <int NT>
__global__ void __launch_bounds__(32) k_wmma_gemm(const unsigned short* __restrict__ A,
                                                  const unsigned short* __restrict__ Bp,
                                                  float* __restrict__ C,
                                                  int K, int Ncols) {
    const int mt     = blockIdx.x;
    const int ntBase = blockIdx.y * NT;
    const int lane   = threadIdx.x;
    const int half   = lane >> 4;      // 0: K 0-15 of chunk, 1: K 16-31
    const int l16    = lane & 15;

    v8f acc[NT];
#pragma unroll
    for (int j = 0; j < NT; ++j) acc[j] = v8f{};

    const int arow = mt * 16 + l16;
    const unsigned short* aBase = A + (size_t)arow * K + half * 16;

    const int ksteps = K >> 5;
    for (int kt = 0; kt < ksteps; ++kt) {
        v16bf a = *reinterpret_cast<const v16bf*>(aBase + kt * 32);
        const int kb = kt * 2 + half;
        const unsigned short* bRow = Bp + ((size_t)kb * Ncols) * 16;
#pragma unroll
        for (int j = 0; j < NT; ++j) {
            const int col = (ntBase + j) * 16 + l16;
            v16bf b = *reinterpret_cast<const v16bf*>(bRow + (size_t)col * 16);
            acc[j] = __builtin_amdgcn_wmma_f32_16x16x32_bf16(false, a, false, b,
                                                             (short)0, acc[j], false, false);
        }
    }
    // D layout: VGPR v -> row mt*16 + v + 8*half, column (ntBase+j)*16 + l16
    const int rbase = mt * 16 + 8 * half;
#pragma unroll
    for (int j = 0; j < NT; ++j) {
        const int col = (ntBase + j) * 16 + l16;
#pragma unroll
        for (int v = 0; v < 8; ++v)
            C[(size_t)(rbase + v) * Ncols + col] = acc[j][v];
    }
}

// al_s[n,h] = dot(h[n,h,:], a_src[h,:]) ; al_d likewise
__global__ void k_logits(const float* __restrict__ Hf,
                         const float* __restrict__ a_src, const float* __restrict__ a_dst,
                         float* __restrict__ als, float* __restrict__ ald, int H, int C) {
    int t = blockIdx.x * blockDim.x + threadIdx.x;
    if (t >= NN * H) return;
    int h = t % H;
    const float* hv = Hf + (size_t)t * C;
    const float* as = a_src + (size_t)h * C;
    const float* ad = a_dst + (size_t)h * C;
    float ss = 0.0f, sd = 0.0f;
    for (int c = 0; c < C; ++c) { float v = hv[c]; ss += v * as[c]; sd += v * ad[c]; }
    als[t] = ss; ald[t] = sd;
}

// zero aggregate buffer, init segment-max encoding and denominators
__global__ void k_init(float* __restrict__ agg, unsigned int* __restrict__ menc,
                       float* __restrict__ den, int nAgg, int nNH) {
    int t = blockIdx.x * blockDim.x + threadIdx.x;
    if (t < nAgg) agg[t] = 0.0f;
    if (t < nNH) { menc[t] = ENC_NEG_INF; den[t] = 0.0f; }
}

// pass A: segment max of leaky-relu logits over incoming edges
__global__ void k_edge_max(const int* __restrict__ ei, const float* __restrict__ als,
                           const float* __restrict__ ald, unsigned int* __restrict__ menc,
                           int H, int ET) {
    int t = blockIdx.x * blockDim.x + threadIdx.x;
    if (t >= ET * H) return;
    int e = t / H, h = t % H, s, d;
    edge_sd(ei, e, s, d);
    float ev = lrelu(als[s * H + h] + ald[d * H + h]);
    atomicMax(&menc[d * H + h], enc_f(ev));
}

// pass B: denom[d,h] += exp(e - m[d,h])
__global__ void k_edge_den(const int* __restrict__ ei, const float* __restrict__ als,
                           const float* __restrict__ ald, const unsigned int* __restrict__ menc,
                           float* __restrict__ den, int H, int ET) {
    int t = blockIdx.x * blockDim.x + threadIdx.x;
    if (t >= ET * H) return;
    int e = t / H, h = t % H, s, d;
    edge_sd(ei, e, s, d);
    float ev = lrelu(als[s * H + h] + ald[d * H + h]);
    float m = dec_f(menc[d * H + h]);
    if (m < -3.0e38f) m = 0.0f;
    atomicAdd(&den[d * H + h], __expf(ev - m));
}

// pass C: agg[d,h,:] += alpha * h[s,h,:]   (one block handles 256/(H*C) edges)
__global__ void __launch_bounds__(256) k_edge_scatter(const int* __restrict__ ei,
                         const float* __restrict__ als, const float* __restrict__ ald,
                         const unsigned int* __restrict__ menc, const float* __restrict__ den,
                         const float* __restrict__ Hf, float* __restrict__ agg,
                         int H, int C, int ET) {
    const int tpe  = H * C;                 // threads per edge (256 or 16)
    const int eper = 256 / tpe;
    int e = blockIdx.x * eper + threadIdx.x / tpe;
    if (e >= ET) return;
    int r = threadIdx.x % tpe;
    int h = r / C, c = r % C, s, d;
    edge_sd(ei, e, s, d);
    float ev = lrelu(als[s * H + h] + ald[d * H + h]);
    float m = dec_f(menc[d * H + h]);
    if (m < -3.0e38f) m = 0.0f;
    float p = __expf(ev - m);
    float alpha = p / (den[d * H + h] + 1e-16f);
    float v = alpha * Hf[((size_t)s * H + h) * C + c];
    atomicAdd(&agg[((size_t)d * H + h) * C + c], v);
}

// head-mean + bias (+ELU, bf16 out for next GEMM) or final f32 output
__global__ void k_final(const float* __restrict__ agg, const float* __restrict__ bias,
                        unsigned short* __restrict__ nextA, float* __restrict__ outF,
                        int H, int C, int last) {
    int t = blockIdx.x * blockDim.x + threadIdx.x;
    if (t >= NN * C) return;
    int n = t / C, c = t % C;
    float ssum = 0.0f;
    for (int h = 0; h < H; ++h) ssum += agg[((size_t)n * H + h) * C + c];
    float y = ssum / (float)H + bias[c];
    if (last) {
        outF[t] = y;
    } else {
        y = y > 0.0f ? y : (__expf(y) - 1.0f);   // ELU
        nextA[t] = f32_to_bf16_rne(y);
    }
}

// ---------------- workspace layout (bytes, 256-aligned) ----------------
#define OFF_AB16   ((size_t)0)                         // bf16 A: 50000*128*2
#define OFF_BPACK  (OFF_AB16  + (size_t)NN*INDIM*2)    // 65536
#define OFF_HFEAT  (OFF_BPACK + (size_t)65536)         // 50000*256*4
#define OFF_ALS    (OFF_HFEAT + (size_t)NN*256*4)
#define OFF_ALD    (OFF_ALS   + (size_t)NN*NHEADS*4)
#define OFF_MENC   (OFF_ALD   + (size_t)NN*NHEADS*4)
#define OFF_DEN    (OFF_MENC  + (size_t)NN*NHEADS*4)
#define OFF_AGG    (OFF_DEN   + (size_t)NN*NHEADS*4)   // 50000*256*4

extern "C" void kernel_launch(void* const* d_in, const int* in_sizes, int n_in,
                              void* d_out, int out_size, void* d_ws, size_t ws_size,
                              hipStream_t stream) {
    (void)in_sizes; (void)n_in; (void)out_size; (void)ws_size;

    const float* x  = (const float*)d_in[0];
    const int*   ei = (const int*)d_in[1];
    const float* W[3]    = { (const float*)d_in[2], (const float*)d_in[6],  (const float*)d_in[10] };
    const float* aS[3]   = { (const float*)d_in[3], (const float*)d_in[7],  (const float*)d_in[11] };
    const float* aD[3]   = { (const float*)d_in[4], (const float*)d_in[8],  (const float*)d_in[12] };
    const float* bias[3] = { (const float*)d_in[5], (const float*)d_in[9],  (const float*)d_in[13] };

    char* ws = (char*)d_ws;
    unsigned short* AB16  = (unsigned short*)(ws + OFF_AB16);
    unsigned short* BPACK = (unsigned short*)(ws + OFF_BPACK);
    float*          HFEAT = (float*)(ws + OFF_HFEAT);
    float*          ALS   = (float*)(ws + OFF_ALS);
    float*          ALD   = (float*)(ws + OFF_ALD);
    unsigned int*   MENC  = (unsigned int*)(ws + OFF_MENC);
    float*          DEN   = (float*)(ws + OFF_DEN);
    float*          AGG   = (float*)(ws + OFF_AGG);
    float*          OUTF  = (float*)d_out;

    const int ET = EE + NN;   // edges + self-loops

    const int Kdim[3]  = { INDIM, HIDD, HIDD };
    const int Hh[3]    = { NHEADS, NHEADS, 1 };
    const int Cc[3]    = { HIDD, HIDD, NCLS };

    // layer 1 A-operand: convert x to bf16
    k_f32_to_bf16<<<(NN * INDIM + 255) / 256, 256, 0, stream>>>(x, AB16, NN * INDIM);

    for (int li = 0; li < 3; ++li) {
        const int K = Kdim[li], H = Hh[li], C = Cc[li];
        const int Ncols = H * C;                      // 256, 256, 16
        const int last  = (li == 2);

        // pack W into WMMA B layout
        int packT = (K / 16) * Ncols;
        k_pack_B<<<(packT + 255) / 256, 256, 0, stream>>>(W[li], BPACK, K, Ncols);

        // GEMM: Hfeat = A(bf16) * W(bf16), f32 accumulate
        if (Ncols == 256) {
            dim3 gg(NN / 16, Ncols / 64);             // 4 N-tiles per wave
            k_wmma_gemm<4><<<gg, 32, 0, stream>>>(AB16, BPACK, HFEAT, K, Ncols);
        } else {
            dim3 gg(NN / 16, Ncols / 16);             // single N-tile (layer 3)
            k_wmma_gemm<1><<<gg, 32, 0, stream>>>(AB16, BPACK, HFEAT, K, Ncols);
        }

        // attention logits
        k_logits<<<(NN * H + 255) / 256, 256, 0, stream>>>(HFEAT, aS[li], aD[li], ALS, ALD, H, C);

        // init segment buffers
        k_init<<<(NN * Ncols + 255) / 256, 256, 0, stream>>>(AGG, MENC, DEN, NN * Ncols, NN * H);

        // edge softmax (max, denom) + scatter aggregate
        k_edge_max<<<(ET * H + 255) / 256, 256, 0, stream>>>(ei, ALS, ALD, MENC, H, ET);
        k_edge_den<<<(ET * H + 255) / 256, 256, 0, stream>>>(ei, ALS, ALD, MENC, DEN, H, ET);
        int eper = 256 / (H * C);
        k_edge_scatter<<<(ET + eper - 1) / eper, 256, 0, stream>>>(ei, ALS, ALD, MENC, DEN,
                                                                   HFEAT, AGG, H, C, ET);

        // head mean + bias (+ ELU and bf16 re-emit for next layer, or final f32 out)
        k_final<<<(NN * C + 255) / 256, 256, 0, stream>>>(AGG, bias[li], AB16,
                                                          last ? OUTF : nullptr, H, C, last);
    }
}
